// PointnetFPModule_5669356832928
// MI455X (gfx1250) — compile-verified, matrix operand order checked by reading, not run on hardware
//
#include <hip/hip_runtime.h>
#include <stdint.h>

// Problem constants (from reference)
#define B_   4
#define N_   8192
#define M_   2048
#define C1_  128
#define C2_  256
#define K0_  384        // C1+C2
#define RO_  256        // output channels of both MLP layers
#define NC_  (B_*N_)    // 32768 total points/columns

// ---------------- CDNA5 async global->LDS copy (probe-guarded) -------------
#if defined(__has_builtin)
# if __has_builtin(__builtin_amdgcn_global_load_async_to_lds_b128)
#  define HAVE_ASYNC_LDS 1
# endif
#endif
#ifndef HAVE_ASYNC_LDS
# define HAVE_ASYNC_LDS 0
#endif

typedef int i32x4_ __attribute__((vector_size(16)));
typedef __attribute__((address_space(1))) i32x4_* gas_q16;
typedef __attribute__((address_space(3))) i32x4_* las_q16;

__device__ __forceinline__ void async_wait0() {
#if HAVE_ASYNC_LDS
# if __has_builtin(__builtin_amdgcn_s_wait_asynccnt)
  __builtin_amdgcn_s_wait_asynccnt(0);
# else
  asm volatile("s_wait_asynccnt 0" ::: "memory");
# endif
#endif
}

// copy 16 bytes global -> LDS
__device__ __forceinline__ void cp16(uint4* ldst, const uint16_t* gsrc) {
#if HAVE_ASYNC_LDS
  __builtin_amdgcn_global_load_async_to_lds_b128(
      (gas_q16)(gsrc), (las_q16)(ldst), 0, 0);
#else
  *ldst = *(const uint4*)gsrc;
#endif
}

// ---------------- bf16 helpers (raw uint16 storage) ----------------
__device__ __forceinline__ uint16_t f2bf(float f) {
  uint32_t u = __float_as_uint(f);
  u += 0x7FFFu + ((u >> 16) & 1u);       // round-to-nearest-even
  return (uint16_t)(u >> 16);
}
__device__ __forceinline__ float bf2f(uint16_t h) {
  return __uint_as_float(((uint32_t)h) << 16);
}

typedef __attribute__((ext_vector_type(16))) __bf16 v16bf;
typedef __attribute__((ext_vector_type(8)))  float  v8f;

union Frag { v16bf v; uint4 q[2]; };

// ---------------- 1) transpose known_feats (B,C2,M) f32 -> (B,M,C2) bf16 ----
__global__ __launch_bounds__(256) void transpose_kf(
    const float* __restrict__ kf, uint16_t* __restrict__ Ft)
{
  __shared__ float tile[32][33];
  const int b  = blockIdx.z;
  const int m0 = blockIdx.x * 32;
  const int c0 = blockIdx.y * 32;
  const int tx = threadIdx.x;   // 0..31
  const int ty = threadIdx.y;   // 0..7
  for (int r = ty; r < 32; r += 8)
    tile[r][tx] = kf[((size_t)b * C2_ + c0 + r) * M_ + m0 + tx];
  __syncthreads();
  for (int r = ty; r < 32; r += 8)
    Ft[((size_t)b * M_ + m0 + r) * C2_ + c0 + tx] = f2bf(tile[tx][r]);
}

// ---------------- 2) fold BN into conv weight/bias, emit bf16 W ------------
__global__ __launch_bounds__(256) void fold_weights(
    const float* __restrict__ W, const float* __restrict__ bconv,
    const float* __restrict__ g, const float* __restrict__ beta,
    const float* __restrict__ rm, const float* __restrict__ rv,
    uint16_t* __restrict__ Wb, float* __restrict__ biasOut,
    int rows, int cols)
{
  int i = blockIdx.x * 256 + threadIdx.x;
  if (i < rows * cols) {
    int r = i / cols;
    float s = g[r] * rsqrtf(rv[r] + 1e-5f);
    Wb[i] = f2bf(W[i] * s);
  }
  if (i < rows) {
    float s = g[i] * rsqrtf(rv[i] + 1e-5f);
    biasOut[i] = (bconv[i] - rm[i]) * s + beta[i];
  }
}

// ---------------- 3) three_nn + three_interpolate + concat -----------------
// Writes X0 point-major bf16: X0T[j][k], j = b*N+n, k in [0,384)
__global__ __launch_bounds__(256) void threenn_interp(
    const float* __restrict__ unknown, const float* __restrict__ known,
    const float* __restrict__ ufeat,  const uint16_t* __restrict__ Ft,
    uint16_t* __restrict__ X0T)
{
  __shared__ float kx[M_], ky[M_], kz[M_];   // 24 KB
  __shared__ int   sIdx[256][3];
  __shared__ float sWt[256][3];
  const int tid = threadIdx.x;
  const int b   = blockIdx.y;
  const int n0  = blockIdx.x * 256;

  const float* kb = known + (size_t)b * M_ * 3;
  for (int i = tid; i < M_; i += 256) {
    kx[i] = kb[i*3+0]; ky[i] = kb[i*3+1]; kz[i] = kb[i*3+2];
  }
  __syncthreads();

  { // phase 1: per-thread 3-NN over LDS-resident known points
    const int n = n0 + tid;
    const float* up = unknown + ((size_t)b * N_ + n) * 3;
    const float ux = up[0], uy = up[1], uz = up[2];
    float b0 = 3.4e38f, b1 = 3.4e38f, b2 = 3.4e38f;
    int   i0 = 0, i1 = 0, i2 = 0;
    for (int m = 0; m < M_; m++) {
      const float dx = ux - kx[m], dy = uy - ky[m], dz = uz - kz[m];
      const float d = dx*dx + dy*dy + dz*dz;
      if (d < b2) {
        if (d < b1) {
          b2 = b1; i2 = i1;
          if (d < b0) { b1 = b0; i1 = i0; b0 = d; i0 = m; }
          else        { b1 = d;  i1 = m; }
        } else        { b2 = d;  i2 = m; }
      }
    }
    const float r0 = 1.f / (sqrtf(b0) + 1e-8f);
    const float r1 = 1.f / (sqrtf(b1) + 1e-8f);
    const float r2 = 1.f / (sqrtf(b2) + 1e-8f);
    const float rs = 1.f / (r0 + r1 + r2);
    sIdx[tid][0] = i0; sIdx[tid][1] = i1; sIdx[tid][2] = i2;
    sWt[tid][0] = r0 * rs; sWt[tid][1] = r1 * rs; sWt[tid][2] = r2 * rs;
  }
  __syncthreads();

  // phase 2: cooperative gather + weighted sum + concat, write point-major bf16
  const int lane = tid & 31;
  const int wave = tid >> 5;
  const uint16_t* Fb = Ft + (size_t)b * M_ * C2_;
  for (int pp = 0; pp < 32; pp++) {
    const int p = wave * 32 + pp;
    const int n = n0 + p;
    const size_t jg = (size_t)b * N_ + n;
    const int m0 = sIdx[p][0], m1 = sIdx[p][1], m2 = sIdx[p][2];
    const float w0 = sWt[p][0], w1 = sWt[p][1], w2 = sWt[p][2];
    uint16_t* xr = X0T + jg * (size_t)K0_;
    const uint16_t* f0 = Fb + (size_t)m0 * C2_;
    const uint16_t* f1 = Fb + (size_t)m1 * C2_;
    const uint16_t* f2 = Fb + (size_t)m2 * C2_;
    #pragma unroll
    for (int c = lane; c < C2_; c += 32) {
      const float v = w0 * bf2f(f0[c]) + w1 * bf2f(f1[c]) + w2 * bf2f(f2[c]);
      xr[c] = f2bf(v);
    }
    #pragma unroll
    for (int cl = lane; cl < C1_; cl += 32) {
      xr[C2_ + cl] = f2bf(ufeat[((size_t)b * C1_ + cl) * N_ + n]);
    }
  }
}

// ---------------- 4) WMMA GEMM: Y = ReLU(Wb @ X + bias) --------------------
// Wb: (256 x K) bf16 row-major. Xin: (NC x K) bf16 point-major.
// FINAL: write (B,256,N) f32; else point-major bf16 (NC x 256).
// Block: 256 thr = 8 waves; block tile 128 rows x 128 cols; wave tile 64x32.
// Double-buffered LDS, async global->LDS staging when available.
template<int K, bool FINAL>
__global__ __launch_bounds__(256) void gemm_bias_relu(
    const uint16_t* __restrict__ Wb, const float* __restrict__ bias,
    const uint16_t* __restrict__ Xin,
    uint16_t* __restrict__ Ybf, float* __restrict__ Yf32)
{
  __shared__ uint16_t sW[2][128 * 32];   // 2 x 8 KB  W tile (rows x k)
  __shared__ uint16_t sX[2][128 * 32];   // 2 x 8 KB  X tile (cols x k)
  const int tid      = threadIdx.x;
  const int lane     = tid & 31;
  const int wave     = tid >> 5;
  const int waveRow  = wave >> 2;     // 0..1  -> 64-row stripe
  const int waveCol  = wave & 3;      // 0..3  -> 32-col stripe
  const int laneHalf = lane >> 4;     // 0/1
  const int lmod     = lane & 15;
  const int rowBase  = blockIdx.y * 128;
  const int colBase  = blockIdx.x * 128;

  // per-thread staging pointers (chunk = 16B; 2 chunks per tile per thread)
  const int srow = tid >> 2;          // 0..63
  const int sseg = (tid & 3) * 8;     // k element offset of chunk
  const uint16_t* wp0 = Wb  + (size_t)(rowBase + srow) * K + sseg;
  const uint16_t* wp1 = wp0 + (size_t)64 * K;
  const uint16_t* xp0 = Xin + (size_t)(colBase + srow) * K + sseg;
  const uint16_t* xp1 = xp0 + (size_t)64 * K;

  auto stage = [&](int buf, int kk) {
    uint4* wq = (uint4*)sW[buf];
    uint4* xq = (uint4*)sX[buf];
    cp16(&wq[tid],       wp0 + kk);
    cp16(&wq[tid + 256], wp1 + kk);
    cp16(&xq[tid],       xp0 + kk);
    cp16(&xq[tid + 256], xp1 + kk);
  };

  v8f acc[4][2];
  const v8f vzero = {0.f,0.f,0.f,0.f,0.f,0.f,0.f,0.f};
  #pragma unroll
  for (int i = 0; i < 4; i++)
    #pragma unroll
    for (int j = 0; j < 2; j++) acc[i][j] = vzero;

  constexpr int NK = K / 32;
  stage(0, 0);
  int cur = 0;
  #pragma unroll
  for (int s = 0; s < NK; s++) {
    async_wait0();        // my async chunks for buffer `cur` have landed
    __syncthreads();      // everyone's chunks have landed; prev reads done
    if (s + 1 < NK) stage(cur ^ 1, (s + 1) * 32);

    Frag a[4], bfr[2];
    #pragma unroll
    for (int fi = 0; fi < 4; fi++) {
      const uint4* p = (const uint4*)&sW[cur][(waveRow*64 + fi*16 + lmod) * 32];
      a[fi].q[0] = p[laneHalf];        // K = laneHalf*8 .. +7
      a[fi].q[1] = p[2 + laneHalf];    // K = 16 + laneHalf*8 .. +7
    }
    #pragma unroll
    for (int fj = 0; fj < 2; fj++) {
      const uint4* p = (const uint4*)&sX[cur][(waveCol*32 + fj*16 + lmod) * 32];
      bfr[fj].q[0] = p[laneHalf];
      bfr[fj].q[1] = p[2 + laneHalf];
    }
    #pragma unroll
    for (int fi = 0; fi < 4; fi++)
      #pragma unroll
      for (int fj = 0; fj < 2; fj++)
        acc[fi][fj] = __builtin_amdgcn_wmma_f32_16x16x32_bf16(
            false, a[fi].v, false, bfr[fj].v, (short)0, acc[fi][fj],
            false, false);
    cur ^= 1;
  }

  // epilogue: bias + ReLU; C layout: lane=col, VGPR v = row (+8 for hi half)
  #pragma unroll
  for (int fi = 0; fi < 4; fi++) {
    const int gm0 = rowBase + waveRow*64 + fi*16 + laneHalf*8;
    float bv[8];
    #pragma unroll
    for (int v = 0; v < 8; v++) bv[v] = bias[gm0 + v];
    #pragma unroll
    for (int fj = 0; fj < 2; fj++) {
      const int jg = colBase + waveCol*32 + fj*16 + lmod;
      float y[8];
      #pragma unroll
      for (int v = 0; v < 8; v++) y[v] = fmaxf(acc[fi][fj][v] + bv[v], 0.f);
      if (FINAL) {       // final output: (B,256,N) f32
        const int bb = jg >> 13;           // / N_
        const int n  = jg & (N_ - 1);
        #pragma unroll
        for (int v = 0; v < 8; v++)
          Yf32[((size_t)bb * RO_ + (gm0 + v)) * N_ + n] = y[v];
      } else {           // intermediate: point-major bf16, 16B per lane
        uint32_t pk[4];
        #pragma unroll
        for (int v = 0; v < 4; v++)
          pk[v] = (uint32_t)f2bf(y[2*v]) | ((uint32_t)f2bf(y[2*v+1]) << 16);
        *(uint4*)(Ybf + (size_t)jg * RO_ + gm0) =
            make_uint4(pk[0], pk[1], pk[2], pk[3]);
      }
    }
  }
}

// ---------------- launch ---------------------------------------------------
extern "C" void kernel_launch(void* const* d_in, const int* in_sizes, int n_in,
                              void* d_out, int out_size, void* d_ws, size_t ws_size,
                              hipStream_t stream) {
  const float* unknown = (const float*)d_in[0];
  const float* known   = (const float*)d_in[1];
  const float* ufeat   = (const float*)d_in[2];
  const float* kfeat   = (const float*)d_in[3];
  const float* W0 = (const float*)d_in[4];
  const float* b0 = (const float*)d_in[5];
  const float* g0 = (const float*)d_in[6];
  const float* be0= (const float*)d_in[7];
  const float* rm0= (const float*)d_in[8];
  const float* rv0= (const float*)d_in[9];
  const float* W1 = (const float*)d_in[10];
  const float* b1 = (const float*)d_in[11];
  const float* g1 = (const float*)d_in[12];
  const float* be1= (const float*)d_in[13];
  const float* rm1= (const float*)d_in[14];
  const float* rv1= (const float*)d_in[15];

  char* ws = (char*)d_ws;
  size_t off = 0;
  auto alloc = [&](size_t bytes) -> void* {
    void* p = ws + off;
    off = (off + bytes + 255) & ~(size_t)255;
    return p;
  };
  uint16_t* Ft    = (uint16_t*)alloc((size_t)B_ * M_ * C2_ * 2);   //  4 MB
  uint16_t* X0T   = (uint16_t*)alloc((size_t)NC_ * K0_ * 2);       // 24 MB
  uint16_t* X1T   = (uint16_t*)alloc((size_t)NC_ * RO_ * 2);       // 16 MB
  uint16_t* Wb0   = (uint16_t*)alloc((size_t)RO_ * K0_ * 2);
  uint16_t* Wb1   = (uint16_t*)alloc((size_t)RO_ * RO_ * 2);
  float*    bias0 = (float*)alloc(RO_ * 4);
  float*    bias1 = (float*)alloc(RO_ * 4);

  transpose_kf<<<dim3(M_/32, C2_/32, B_), dim3(32, 8), 0, stream>>>(kfeat, Ft);
  fold_weights<<<(RO_*K0_ + 255)/256, 256, 0, stream>>>(
      W0, b0, g0, be0, rm0, rv0, Wb0, bias0, RO_, K0_);
  fold_weights<<<(RO_*RO_ + 255)/256, 256, 0, stream>>>(
      W1, b1, g1, be1, rm1, rv1, Wb1, bias1, RO_, RO_);
  threenn_interp<<<dim3(N_/256, B_), 256, 0, stream>>>(
      unknown, known, ufeat, Ft, X0T);
  gemm_bias_relu<K0_, false><<<dim3(NC_/128, RO_/128), 256, 0, stream>>>(
      Wb0, bias0, X0T, X1T, nullptr);
  gemm_bias_relu<RO_, true><<<dim3(NC_/128, RO_/128), 256, 0, stream>>>(
      Wb1, bias1, X1T, nullptr, (float*)d_out);
}
